// CodebookSampler_10634339025302
// MI455X (gfx1250) — matrix operand order repaired and checked by reading, not compiled, not run on hardware
//
#include <hip/hip_runtime.h>
#include <hip/hip_bf16.h>
#include <stdint.h>

// ---------------------------------------------------------------------------
// CodebookSampler for MI455X (gfx1250, wave32)
//   scores  s[b,c,t] = |x_t|^2 - 2 * dot(codebook_c, x_bt)   (|c|^2 dropped:
//                     constant along the argmin-over-t axis)
//   GEMM    256 x 65536 x 1024 via v_wmma_f32_16x16x32_bf16 (f32 accum)
//           block = 32 codes x 128 tokens; 2 WMMA tiles share one B fragment
//           codebook tile staged in LDS via global_load_async_to_lds_b128
//   argmin  packed (sortable-f32-key | token) u64, shfl + LDS + atomicMin
//   output  ste = gathered rows, loss = 2*mean((g - cb)^2)
// ---------------------------------------------------------------------------

typedef __attribute__((ext_vector_type(16))) __bf16        v16bf;
typedef __attribute__((ext_vector_type(8)))  float         v8f;
typedef __attribute__((ext_vector_type(8)))  unsigned int  v8u;

constexpr int NC = 256;    // codes
constexpr int FD = 1024;   // feature dim (K)
constexpr int NB = 16;     // batch
constexpr int NT = 4096;   // tokens per batch

constexpr int CPB    = 32;               // codes per block (2 WMMA M tiles)
constexpr int TWAVE  = 16;               // tokens per wave (one WMMA N tile)
constexpr int NWAVES = 8;                // waves per block (256 threads)
constexpr int TBLOCK = TWAVE * NWAVES;   // 128 tokens per block
constexpr int LDSW   = (FD / 2) + 4;     // uint stride per cb row (pad -> 4-bank shift)

// float -> monotonically ordered u32 key (ascending uint == ascending float)
__device__ __forceinline__ unsigned int fkey(float f) {
  unsigned int u = __builtin_bit_cast(unsigned int, f);
  return (u & 0x80000000u) ? ~u : (u | 0x80000000u);
}

// two f32 -> packed bf16x2 by byte-permute (truncation) : ONE v_perm_b32
__device__ __forceinline__ unsigned int pkbf2(float lo, float hi) {
  return __builtin_amdgcn_perm(__builtin_bit_cast(unsigned int, hi),
                               __builtin_bit_cast(unsigned int, lo),
                               0x07060302u);
}

__device__ __forceinline__ unsigned long long shfl_xor_u64(unsigned long long v, int mask) {
  unsigned int lo = (unsigned int)v;
  unsigned int hi = (unsigned int)(v >> 32);
  lo = __shfl_xor(lo, mask, 32);
  hi = __shfl_xor(hi, mask, 32);
  return ((unsigned long long)hi << 32) | lo;
}

// ---------------------------------------------------------------- kernel 1 --
__global__ void cbs_init(unsigned long long* __restrict__ amin,
                         float* __restrict__ lossp) {
  int i = blockIdx.x * blockDim.x + threadIdx.x;
  if (i < NB * NC) amin[i] = 0xFFFFFFFFFFFFFFFFull;
  if (i == 0) *lossp = 0.0f;
}

// ---------------------------------------------------------------- kernel 2 --
// codebook f32 -> bf16x2 copy in workspace (one block per code row)
__global__ void cbs_cvt_cb(const float* __restrict__ cb, unsigned int* __restrict__ cb16) {
  const int row = blockIdx.x;
  for (int i = threadIdx.x; i < FD / 2; i += blockDim.x) {
    float2 v = *(const float2*)(cb + (size_t)row * FD + 2 * i);
    cb16[(size_t)row * (FD / 2) + i] = pkbf2(v.x, v.y);
  }
}

// ---------------------------------------------------------------- kernel 3 --
// one wave32 per token: xnorm[b*NT+t] = sum_d x^2
__global__ void cbs_xnorm(const float* __restrict__ x, float* __restrict__ xnorm) {
  const int wave = (blockIdx.x * blockDim.x + threadIdx.x) >> 5;
  const int lane = threadIdx.x & 31;
  const float* xp = x + (size_t)wave * FD + lane * 4;
  float s = 0.0f;
#pragma unroll
  for (int j = 0; j < 8; ++j) {
    float4 v = *(const float4*)(xp + j * 128);
    s = fmaf(v.x, v.x, s);
    s = fmaf(v.y, v.y, s);
    s = fmaf(v.z, v.z, s);
    s = fmaf(v.w, v.w, s);
  }
#pragma unroll
  for (int off = 16; off; off >>= 1) s += __shfl_xor(s, off, 32);
  if (lane == 0) xnorm[wave] = s;
}

// ---------------------------------------------------------------- kernel 4 --
// fused bf16 WMMA GEMM + argmin.  grid = (NT/128, NC/32, NB), block = 256.
__global__ void __launch_bounds__(256, 2)
cbs_score_argmin(const float* __restrict__ x,
                 const unsigned int* __restrict__ cb16,
                 const float* __restrict__ xnorm,
                 unsigned long long* __restrict__ amin) {
  __shared__ unsigned int        lds_cb[CPB][LDSW];     // 32 code rows, bf16x2
  __shared__ unsigned long long  lds_min[CPB][NWAVES];  // per-wave argmin keys

  const int b     = blockIdx.z;
  const int cbase = blockIdx.y * CPB;
  const int tbase = blockIdx.x * TBLOCK;
  const int tid   = threadIdx.x;
  const int wave  = tid >> 5;
  const int lane  = tid & 31;
  const int laneN = lane & 15;
  const int hi    = lane >> 4;          // lane half: K-group selector

  // --- stage codebook tile (32 rows x 2048B) into LDS via async DMA -------
  // 16-byte chunks; dsaddr = low 32 bits of the flat shared-pointer (LDS offset)
  for (int chunk = tid; chunk < CPB * (FD / 8); chunk += 256) {
    const int row = chunk >> 7;          // FD/8 = 128 chunks of 16B per row
    const int col = chunk & 127;
    unsigned int ldsa = (unsigned int)(size_t)&lds_cb[row][col * 4];
    const unsigned int* gsrc = cb16 + (size_t)(cbase + row) * (FD / 2) + col * 4;
    asm volatile("global_load_async_to_lds_b128 %0, %1, off"
                 :: "v"(ldsa), "v"(gsrc) : "memory");
  }
  asm volatile("s_wait_asynccnt 0x0" ::: "memory");
  __syncthreads();

  const int tok = tbase + wave * TWAVE + laneN;           // this lane's token col
  const float* xrow = x + ((size_t)b * NT + tok) * FD + hi * 16;

  v8f acc0 = {0.f, 0.f, 0.f, 0.f, 0.f, 0.f, 0.f, 0.f};
  v8f acc1 = acc0;

#pragma unroll 4
  for (int k0 = 0; k0 < FD; k0 += 32) {
    // B fragment (32x16 bf16): lane<16 -> N=tok, K k0..k0+15;
    //                          lane>=16 -> N=tok, K k0+16..k0+31 (xrow pre-offset)
    float4 f0 = *(const float4*)(xrow + k0 + 0);
    float4 f1 = *(const float4*)(xrow + k0 + 4);
    float4 f2 = *(const float4*)(xrow + k0 + 8);
    float4 f3 = *(const float4*)(xrow + k0 + 12);
    v8u bu;
    bu[0] = pkbf2(f0.x, f0.y); bu[1] = pkbf2(f0.z, f0.w);
    bu[2] = pkbf2(f1.x, f1.y); bu[3] = pkbf2(f1.z, f1.w);
    bu[4] = pkbf2(f2.x, f2.y); bu[5] = pkbf2(f2.z, f2.w);
    bu[6] = pkbf2(f3.x, f3.y); bu[7] = pkbf2(f3.z, f3.w);
    v16bf Bm = __builtin_bit_cast(v16bf, bu);

    // A fragments (16x32 bf16): lane<16 -> M=laneN, K k0..k0+15;
    //                           lane>=16 -> same M, K k0+16..k0+31
    const int g1 = (k0 >> 1) + hi * 8;
    {
      uint4 q1 = *(const uint4*)&lds_cb[laneN][g1];
      uint4 q2 = *(const uint4*)&lds_cb[laneN][g1 + 4];
      v8u au;
      au[0] = q1.x; au[1] = q1.y; au[2] = q1.z; au[3] = q1.w;
      au[4] = q2.x; au[5] = q2.y; au[6] = q2.z; au[7] = q2.w;
      v16bf A0 = __builtin_bit_cast(v16bf, au);
      acc0 = __builtin_amdgcn_wmma_f32_16x16x32_bf16(false, A0, false, Bm,
                                                     (short)0, acc0, false, false);
    }
    {
      uint4 q1 = *(const uint4*)&lds_cb[16 + laneN][g1];
      uint4 q2 = *(const uint4*)&lds_cb[16 + laneN][g1 + 4];
      v8u au;
      au[0] = q1.x; au[1] = q1.y; au[2] = q1.z; au[3] = q1.w;
      au[4] = q2.x; au[5] = q2.y; au[6] = q2.z; au[7] = q2.w;
      v16bf A1 = __builtin_bit_cast(v16bf, au);
      acc1 = __builtin_amdgcn_wmma_f32_16x16x32_bf16(false, A1, false, Bm,
                                                     (short)0, acc1, false, false);
    }
  }

  const float xn = xnorm[(size_t)b * NT + tok];

  // C/D layout: VGPR i holds (M=i, N=laneN) on lanes 0-15, (M=i+8, N=laneN) on 16-31
#pragma unroll
  for (int ct = 0; ct < 2; ++ct) {
    v8f acc = ct ? acc1 : acc0;
#pragma unroll
    for (int i = 0; i < 8; ++i) {
      float s = fmaf(-2.0f, acc[i], xn);
      unsigned long long key = ((unsigned long long)fkey(s) << 32) | (unsigned int)tok;
#pragma unroll
      for (int off = 1; off < 16; off <<= 1) {  // min across 16 tokens (stays in half)
        unsigned long long o = shfl_xor_u64(key, off);
        key = (o < key) ? o : key;
      }
      if (laneN == 0) lds_min[ct * 16 + i + hi * 8][wave] = key;
    }
  }
  __syncthreads();

  if (tid < CPB) {
    unsigned long long best = lds_min[tid][0];
#pragma unroll
    for (int w = 1; w < NWAVES; ++w) {
      unsigned long long o = lds_min[tid][w];
      best = (o < best) ? o : best;
    }
    atomicMin(&amin[(size_t)b * NC + cbase + tid], best);
  }
}

// ---------------------------------------------------------------- kernel 5 --
// one block per (b,c): gather winning row, accumulate loss
__global__ void cbs_gather_loss(const float* __restrict__ x,
                                const float* __restrict__ cb,
                                const unsigned long long* __restrict__ amin,
                                float* __restrict__ ste,
                                float* __restrict__ lossp) {
  __shared__ float red[NWAVES];
  const int blk = blockIdx.x;            // = b*NC + c
  const int c   = blk & (NC - 1);
  const int b   = blk >> 8;
  const unsigned int idx = (unsigned int)(amin[blk] & 0xFFFFFFFFull);

  const float* xg = x + ((size_t)b * NT + idx) * FD;
  const float* cr = cb + (size_t)c * FD;
  float* og = ste + (size_t)blk * FD;

  const int d = threadIdx.x * 4;         // 256 threads x float4 == 1024
  float4 g = *(const float4*)(xg + d);
  float4 q = *(const float4*)(cr + d);
  *(float4*)(og + d) = g;                // STE forward value == gathered

  float dx = g.x - q.x, dy = g.y - q.y, dz = g.z - q.z, dw = g.w - q.w;
  float s = dx * dx + dy * dy + dz * dz + dw * dw;
#pragma unroll
  for (int off = 16; off; off >>= 1) s += __shfl_xor(s, off, 32);
  if ((threadIdx.x & 31) == 0) red[threadIdx.x >> 5] = s;
  __syncthreads();
  if (threadIdx.x < 8) {
    s = red[threadIdx.x];
#pragma unroll
    for (int off = 4; off; off >>= 1) s += __shfl_xor(s, off, 32);
    if (threadIdx.x == 0)
      atomicAdd(lossp, s * (2.0f / ((float)NB * NC * FD)));
  }
}

// ---------------------------------------------------------------------------
extern "C" void kernel_launch(void* const* d_in, const int* in_sizes, int n_in,
                              void* d_out, int out_size, void* d_ws, size_t ws_size,
                              hipStream_t stream) {
  (void)in_sizes; (void)n_in; (void)ws_size;
  const float* x  = (const float*)d_in[0];   // [16, 4096, 1024] f32
  const float* cb = (const float*)d_in[1];   // [256, 1024] f32
  float* out = (float*)d_out;                // ste [16,256,1024] ++ loss [1]

  // workspace layout: amin (32 KB) | xnorm (256 KB) | cb16 (512 KB)
  unsigned long long* amin = (unsigned long long*)d_ws;
  float* xnorm = (float*)((char*)d_ws + (size_t)NB * NC * sizeof(unsigned long long));
  unsigned int* cb16 = (unsigned int*)((char*)xnorm + (size_t)NB * NT * sizeof(float));
  float* lossp = out + (out_size - 1);

  cbs_init<<<dim3((NB * NC + 255) / 256), 256, 0, stream>>>(amin, lossp);
  cbs_cvt_cb<<<dim3(NC), 256, 0, stream>>>(cb, cb16);
  cbs_xnorm<<<dim3(NB * NT / NWAVES), 256, 0, stream>>>(x, xnorm);
  cbs_score_argmin<<<dim3(NT / TBLOCK, NC / CPB, NB), 256, 0, stream>>>(x, cb16, xnorm, amin);
  cbs_gather_loss<<<dim3(NB * NC), 256, 0, stream>>>(x, cb, amin, out, lossp);
}
